// ECE_94489280550
// MI455X (gfx1250) — matrix operand order repaired and checked by reading, not compiled, not run on hardware
//
#include <hip/hip_runtime.h>

// ECE (expected calibration error) for N floats + N bool flags.
// Bandwidth-bound: 42 MB @ 23.3 TB/s ~ 1.8us floor. Segmented sums are done
// as one-hot matrix products on the WMMA pipe (v_wmma_f32_16x16x32_f16),
// accumulating per-bin {count, sum(conf), sum(correct)} in f32 C registers.

typedef __attribute__((ext_vector_type(16))) _Float16 v16h;
typedef __attribute__((ext_vector_type(8)))  float    v8f;

#define ECE_BINS 20

__global__ void ece_zero_ws(float* ws) {
    if (threadIdx.x < 64) ws[threadIdx.x] = 0.0f;
}

// ws layout: [0..19]=counts  [20..39]=sum(conf)  [40..59]=sum(correct)
__global__ __launch_bounds__(256) void ece_hist_wmma(
        const float* __restrict__ confs,
        const unsigned char* __restrict__ corrects,   // jnp bool -> 1 byte
        float* __restrict__ ws, int n) {
    // Per-wave private staging (8 waves/block, 32 entries each). No cross-wave
    // sharing -> only wave-local LDS ordering (s_wait_dscnt) is required.
    __shared__ _Float16       s_conf[256];
    __shared__ _Float16       s_corr[256];
    __shared__ unsigned short s_bin[256];

    const int lane     = threadIdx.x & 31;
    const int waveBase = threadIdx.x & ~31;          // wave's LDS base
    const int tid      = blockIdx.x * blockDim.x + threadIdx.x;
    const int stride   = gridDim.x * blockDim.x;
    const int nIter    = (n + stride - 1) / stride;  // uniform across grid

    // A-matrix lane mapping (16-bit A 16x32, ISA 7.12.2):
    //   lanes 0-15 : row M=lane,    K = {0..7, 16..23}
    //   lanes16-31 : row M=lane-16, K = {8..15, 24..31}
    const int m     = lane & 15;
    const int khalf = lane >> 4;                     // which K half this lane owns

    v8f c0 = {};   // D accumulator, bins 0..15
    v8f c1 = {};   // D accumulator, bins 16..31 (only 16..19 ever hit)

    for (int t = 0; t < nIter; ++t) {
        const int i = tid + t * stride;
        float cf = 0.0f, cr = 0.0f;
        int   b  = 0x7FFF;                           // matches no one-hot column
        if (i < n) {
            cf = confs[i];
            cr = corrects[i] ? 1.0f : 0.0f;
            b  = (int)(cf * (float)ECE_BINS);        // trunc == floor (cf >= 0)
            b  = b < 0 ? 0 : (b > ECE_BINS - 1 ? ECE_BINS - 1 : b);
        }
        // Stage this wave's 32 elements for cross-lane fragment construction.
        s_conf[threadIdx.x] = (_Float16)cf;
        s_corr[threadIdx.x] = (_Float16)cr;
        s_bin [threadIdx.x] = (unsigned short)b;
        asm volatile("s_wait_dscnt 0" ::: "memory"); // wave-local visibility

        // ---- A (values, 16x32 f16): row0 = 1, row1 = conf, row2 = correct ----
        v16h A = {};
        if (m == 0) {
            #pragma unroll
            for (int e = 0; e < 16; ++e) A[e] = (_Float16)1.0f;
        } else if (m == 1) {
            const _Float16* p = &s_conf[waveBase];
            #pragma unroll
            for (int e = 0; e < 8; ++e) {            // elems 0..7 -> K khalf*8+e
                A[e]     = p[khalf * 8 + e];         // elems 8..15 -> K 16+khalf*8+e
                A[e + 8] = p[16 + khalf * 8 + e];
            }
        } else if (m == 2) {
            const _Float16* p = &s_corr[waveBase];
            #pragma unroll
            for (int e = 0; e < 8; ++e) {
                A[e]     = p[khalf * 8 + e];
                A[e + 8] = p[16 + khalf * 8 + e];
            }
        }

        // ---- B (one-hot, 32x16 f16): col n = lane&15; same K order as A:
        //      elems 0..7 -> K khalf*8+e, elems 8..15 -> K 16+khalf*8+(e-8).
        const unsigned short* pb = &s_bin[waveBase + khalf * 8];
        v16h B0 = {}, B1 = {};
        #pragma unroll
        for (int e = 0; e < 16; ++e) {
            const int bk = pb[e < 8 ? e : e + 8];
            B0[e] = (bk == m)      ? (_Float16)1.0f : (_Float16)0.0f; // bins 0..15
            B1[e] = (bk == m + 16) ? (_Float16)1.0f : (_Float16)0.0f; // bins 16..31
        }

        // D[m][n] += sum_k A[m][k] * onehot[k][n] : per-bin segmented sums.
        c0 = __builtin_amdgcn_wmma_f32_16x16x32_f16(false, A, false, B0,
                                                    (short)0, c0, false, false);
        c1 = __builtin_amdgcn_wmma_f32_16x16x32_f16(false, A, false, B1,
                                                    (short)0, c1, false, false);
    }

    // D layout: VGPR j, lanes 0-15 -> M=j, N=lane. Rows 0/1/2 = cnt/conf/corr.
    // Lanes 16-31 hold rows M=8..15 which are all zero -> skip.
    if (lane < 16) {
        atomicAdd(&ws[ 0 + lane], c0[0]);
        atomicAdd(&ws[20 + lane], c0[1]);
        atomicAdd(&ws[40 + lane], c0[2]);
        if (lane < ECE_BINS - 16) {                  // bins 16..19 from tile 1
            atomicAdd(&ws[ 0 + 16 + lane], c1[0]);
            atomicAdd(&ws[20 + 16 + lane], c1[1]);
            atomicAdd(&ws[40 + 16 + lane], c1[2]);
        }
    }
}

__global__ void ece_finalize(const float* __restrict__ ws, float* __restrict__ out) {
    if (threadIdx.x == 0) {
        const float tiny = 1.17549435e-38f;          // finfo(float32).tiny
        float total = 0.0f, s = 0.0f;
        #pragma unroll
        for (int b = 0; b < ECE_BINS; ++b) total += ws[b];
        #pragma unroll
        for (int b = 0; b < ECE_BINS; ++b) {
            const float cnt = ws[b];
            const float err = fabsf(ws[20 + b] - ws[40 + b]) / (cnt + tiny);
            s += err * cnt;                          // NORM == 1
        }
        out[0] = s / total;
    }
}

extern "C" void kernel_launch(void* const* d_in, const int* in_sizes, int n_in,
                              void* d_out, int out_size, void* d_ws, size_t ws_size,
                              hipStream_t stream) {
    const float*         confs    = (const float*)d_in[0];
    const unsigned char* corrects = (const unsigned char*)d_in[1];  // jnp bool
    float* out = (float*)d_out;
    float* ws  = (float*)d_ws;
    const int n = in_sizes[0];

    // Same stream -> kernels serialize: zero scratch, histogram, finalize.
    ece_zero_ws<<<1, 64, 0, stream>>>(ws);
    const int blocks = 2048;                         // 524288 threads, 16 iters @ 8.4M
    ece_hist_wmma<<<blocks, 256, 0, stream>>>(confs, corrects, ws, n);
    ece_finalize<<<1, 32, 0, stream>>>(ws, out);
}
